// VectorQuantizerLayer_26740466385252
// MI455X (gfx1250) — matrix operand order repaired and checked by reading, not compiled, not run on hardware
//
#include <hip/hip_runtime.h>
#include <hip/hip_bf16.h>

// ---------------------------------------------------------------------------
// VQ-VAE vector quantizer for MI455X (gfx1250, wave32, WMMA).
// inputs  [16,2048,512] f32 ; embeddings [512,1024] f32
// cluster_size_ema [1024]   ; embeddings_ema [512,1024]
// outputs (concat): quantized_st [16M] | loss [1] | new_emb [512K] |
//                   new_cse [1K] | new_emb_ema [512K]
// ---------------------------------------------------------------------------

#define N_ROWS 32768
#define EMB_D  512
#define N_EMB  1024
#define GAMMA  0.99f
#define BETA   0.25f
#define VQ_EPS 1e-5f

typedef __attribute__((ext_vector_type(16))) _Float16 v16h;
typedef __attribute__((ext_vector_type(8)))  float    v8f;

// ---------------------------------------------------------------------------
// Kernel 1: squared norms of codebook columns. enorm[k] = sum_d E[d,k]^2
// ---------------------------------------------------------------------------
__global__ __launch_bounds__(256) void vq_enorm_kernel(
    const float* __restrict__ E, float* __restrict__ enorm) {
  int k = blockIdx.x * 256 + threadIdx.x;       // 0..1023
  float s = 0.0f;
  #pragma unroll 8
  for (int d = 0; d < EMB_D; ++d) {
    float e = E[d * N_EMB + k];
    s += e * e;
  }
  enorm[k] = s;
}

// ---------------------------------------------------------------------------
// Kernel 2: pack codebook into per-fragment WMMA B layout (f16).
// For k-tile kt (64), fragment f (16, d-base 32f), lane l:
//   n = l%16, d = 32f + 16*(l/16) + j  (j = 0..15)
// Bp[((kt*16+f)*32 + l)*16 + j] = (f16) E[d, kt*16+n]
// One v16h (32B) contiguous per lane -> coalesced fragment loads in the GEMM.
// ---------------------------------------------------------------------------
__global__ __launch_bounds__(256) void vq_pack_b_kernel(
    const float* __restrict__ E, _Float16* __restrict__ Bp) {
  int t = blockIdx.x * 256 + threadIdx.x;       // 0 .. 32767
  int l     = t & 31;
  int frag  = (t >> 5) & 15;
  int kt    = t >> 9;
  int n     = l & 15;
  int dbase = 32 * frag + 16 * (l >> 4);
  int k     = kt * 16 + n;
  v16h v;
  #pragma unroll
  for (int j = 0; j < 16; ++j)
    v[j] = (_Float16)E[(dbase + j) * N_EMB + k];
  *(((v16h*)Bp) + t) = v;
}

// ---------------------------------------------------------------------------
// Kernel 3: fused GEMM + argmin.  256 threads = 8 waves; wave w owns rows
// [blk*128 + w*16, +16).  A (16x512 f32) is converted to 16 f16 WMMA
// fragments held in registers.  B tiles (16 KB each, 64 of them) are
// software-pipelined into a double-buffered LDS region with CDNA5
// GLOBAL_LOAD_ASYNC_TO_LDS_B128 (ASYNCcnt) so the copy of tile kt+1 fully
// overlaps the 16 chained v_wmma_f32_16x16x32_f16 of tile kt.
// Score s = ||e_k||^2 - 2*x.e_k ; per-lane running (min, argidx), final
// 16-lane shfl_xor reduction gives argmin per row.
// ---------------------------------------------------------------------------
// Issue one thread's share (4 x 16B) of a 16 KB tile: lds/global base in
// VGPRs, INST_OFFSET is added to BOTH global and LDS addresses by HW.
#define VQ_ASYNC_TILE(ldsaddr, goff, base)                                    \
  do {                                                                        \
    asm volatile("global_load_async_to_lds_b128 %0, %1, %2 offset:0"          \
                 :: "v"(ldsaddr), "v"(goff), "s"(base) : "memory");           \
    asm volatile("global_load_async_to_lds_b128 %0, %1, %2 offset:4096"       \
                 :: "v"(ldsaddr), "v"(goff), "s"(base) : "memory");           \
    asm volatile("global_load_async_to_lds_b128 %0, %1, %2 offset:8192"       \
                 :: "v"(ldsaddr), "v"(goff), "s"(base) : "memory");           \
    asm volatile("global_load_async_to_lds_b128 %0, %1, %2 offset:12288"      \
                 :: "v"(ldsaddr), "v"(goff), "s"(base) : "memory");           \
  } while (0)

__global__ __launch_bounds__(256) void vq_argmin_gemm_kernel(
    const float*    __restrict__ x,       // [N_ROWS, EMB_D]
    const _Float16* __restrict__ Bp,      // packed codebook fragments
    const float*    __restrict__ enorm,   // [N_EMB]
    int*            __restrict__ idx_out) // [N_ROWS]
{
  __shared__ __align__(32) _Float16 ldsB[2 * 16 * 32 * 16];   // 2 x 16 KB

  const int tid  = threadIdx.x;
  const int wave = tid >> 5;
  const int lane = tid & 31;
  const int half = lane >> 4;        // 0: lanes 0-15, 1: lanes 16-31
  const int l16  = lane & 15;
  const int row0 = (blockIdx.x * 8 + wave) * 16;

  const unsigned ldsbase = (unsigned)(uintptr_t)(&ldsB[0]) + (unsigned)tid * 16u;

  // ---- preload A fragments (contiguous k = 16h + s, same order as packed B) ----
  // lane (half h): row m = l16 ; frag f: d in [32f+16h, +16)
  v16h a[16];
  const float* xrow = x + (size_t)(row0 + l16) * EMB_D;
  #pragma unroll
  for (int f = 0; f < 16; ++f) {
    const float* p = xrow + 32 * f + 16 * half;
    v16h af;
    #pragma unroll
    for (int j = 0; j < 8; ++j) {
      af[j]     = (_Float16)p[j];
      af[8 + j] = (_Float16)p[8 + j];
    }
    a[f] = af;
  }

  float minv[8];
  int   mini[8];
  #pragma unroll
  for (int j = 0; j < 8; ++j) { minv[j] = 3.4e38f; mini[j] = 0; }

  // ---- prologue: start async copy of tile 0 into buffer 0 ----
  VQ_ASYNC_TILE(ldsbase, (unsigned)tid * 16u, Bp);

  for (int kt = 0; kt < 64; ++kt) {
    const int cur = kt & 1;

    if (kt + 1 < 64) {
      // kick tile kt+1 into the other buffer, then wait until only those
      // 4 newest per-thread async ops remain outstanding (tile kt landed).
      const unsigned nl = ldsbase + (unsigned)((cur ^ 1) * 16384);
      const unsigned ng = (unsigned)((kt + 1) * 16384) + (unsigned)tid * 16u;
      VQ_ASYNC_TILE(nl, ng, Bp);
      asm volatile("s_wait_asynccnt 0x4" ::: "memory");
    } else {
      asm volatile("s_wait_asynccnt 0x0" ::: "memory");
    }
    __syncthreads();   // all waves' shares of tile kt visible

    const v16h* ldsv = (const v16h*)(ldsB + cur * 8192);
    v8f acc = {};
    #pragma unroll
    for (int f = 0; f < 16; ++f) {
      v16h b = ldsv[f * 32 + lane];
      acc = __builtin_amdgcn_wmma_f32_16x16x32_f16(
          /*neg_a=*/false, a[f], /*neg_b=*/false, b,
          /*c_mod=*/(short)0, acc, /*reuse_a=*/false, /*reuse_b=*/false);
    }

    // distance score for column k = kt*16 + l16 (same k for every C VGPR)
    const int   kk = kt * 16 + l16;
    const float en = enorm[kk];
    #pragma unroll
    for (int j = 0; j < 8; ++j) {
      float s = en - 2.0f * acc[j];
      if (s < minv[j]) { minv[j] = s; mini[j] = kk; }
    }
    __syncthreads();   // reads of buf cur done before it is overwritten
  }

  // ---- argmin reduction across the 16 lanes of each half ----
  // C layout: VGPR j -> row j (lanes 0-15) / row j+8 (lanes 16-31)
  #pragma unroll
  for (int j = 0; j < 8; ++j) {
    float v = minv[j];
    int   i = mini[j];
    #pragma unroll
    for (int off = 8; off > 0; off >>= 1) {
      float ov = __shfl_xor(v, off, 16);
      int   oi = __shfl_xor(i, off, 16);
      if (ov < v || (ov == v && oi < i)) { v = ov; i = oi; }
    }
    if (l16 == 0) idx_out[row0 + 8 * half + j] = i;
  }
}

// ---------------------------------------------------------------------------
// Kernel 4: gather quantized output, per-row loss partial, scatter stats.
// One block per row n; 512 threads = one per feature d.
// ---------------------------------------------------------------------------
__global__ __launch_bounds__(512) void vq_quantize_scatter_kernel(
    const float* __restrict__ x,
    const float* __restrict__ E,
    const int*   __restrict__ idx,
    float* __restrict__ q_out,          // [N_ROWS, EMB_D]
    float* __restrict__ emb_sum,        // [EMB_D, N_EMB] (zeroed)
    float* __restrict__ counts,         // [N_EMB] (zeroed)
    float* __restrict__ loss_partial)   // [N_ROWS]
{
  const int n = blockIdx.x;
  const int d = threadIdx.x;
  const int k = idx[n];

  const float e  = E[d * N_EMB + k];
  const float xv = x[(size_t)n * EMB_D + d];
  q_out[(size_t)n * EMB_D + d] = e;     // quantized_st == quantized

  float s = (e - xv) * (e - xv);

  __shared__ float red[16];
  #pragma unroll
  for (int off = 16; off > 0; off >>= 1) s += __shfl_xor(s, off, 32);
  if ((d & 31) == 0) red[d >> 5] = s;
  __syncthreads();
  if (d < 16) {
    float v = red[d];
    #pragma unroll
    for (int off = 8; off > 0; off >>= 1) v += __shfl_xor(v, off, 16);
    if (d == 0) loss_partial[n] = v;
  }

  atomicAdd(&emb_sum[d * N_EMB + k], xv);
  if (d == 0) atomicAdd(&counts[k], 1.0f);
}

// ---------------------------------------------------------------------------
// Kernel 5: cluster-size EMA, Laplace smoothing, loss finalize. One block.
// ---------------------------------------------------------------------------
__global__ __launch_bounds__(1024) void vq_finalize_kernel(
    const float* __restrict__ counts,
    const float* __restrict__ cse,
    const float* __restrict__ loss_partial,
    float* __restrict__ out_ncse,       // d_out slice [N_EMB]
    float* __restrict__ out_loss,       // d_out slice [1]
    float* __restrict__ cs_ws)          // [N_EMB]
{
  const int t = threadIdx.x, lane = t & 31, wave = t >> 5;
  __shared__ float red[32];

  const float ncse = cse[t] * GAMMA + counts[t] * (1.0f - GAMMA);
  out_ncse[t] = ncse;

  // sum(ncse)
  float s = ncse;
  #pragma unroll
  for (int off = 16; off > 0; off >>= 1) s += __shfl_xor(s, off, 32);
  if (lane == 0) red[wave] = s;
  __syncthreads();
  if (t < 32) {
    float v = red[t];
    #pragma unroll
    for (int off = 16; off > 0; off >>= 1) v += __shfl_xor(v, off, 32);
    if (t == 0) red[0] = v;
  }
  __syncthreads();
  const float nsum = red[0];
  cs_ws[t] = (ncse + VQ_EPS) / (nsum + (float)N_EMB * VQ_EPS) * nsum;
  __syncthreads();   // protect red[] before reuse

  // commitment loss
  float ls = 0.0f;
  for (int i = t; i < N_ROWS; i += 1024) ls += loss_partial[i];
  #pragma unroll
  for (int off = 16; off > 0; off >>= 1) ls += __shfl_xor(ls, off, 32);
  if (lane == 0) red[wave] = ls;
  __syncthreads();
  if (t < 32) {
    float v = red[t];
    #pragma unroll
    for (int off = 16; off > 0; off >>= 1) v += __shfl_xor(v, off, 32);
    if (t == 0)
      out_loss[0] = BETA * v / ((float)N_ROWS * (float)EMB_D);
  }
}

// ---------------------------------------------------------------------------
// Kernel 6: codebook EMA + normalized embedding output.
// ---------------------------------------------------------------------------
__global__ __launch_bounds__(256) void vq_emb_update_kernel(
    const float* __restrict__ emb_ema,
    const float* __restrict__ emb_sum,
    const float* __restrict__ cs,
    float* __restrict__ out_new_emb,    // d_out slice [EMB_D*N_EMB]
    float* __restrict__ out_new_ema)    // d_out slice [EMB_D*N_EMB]
{
  const int i = blockIdx.x * 256 + threadIdx.x;   // < 524288
  const int k = i & (N_EMB - 1);
  const float ema = emb_ema[i] * GAMMA + emb_sum[i] * (1.0f - GAMMA);
  out_new_ema[i] = ema;
  out_new_emb[i] = ema / cs[k];
}

// ---------------------------------------------------------------------------
extern "C" void kernel_launch(void* const* d_in, const int* in_sizes, int n_in,
                              void* d_out, int out_size, void* d_ws, size_t ws_size,
                              hipStream_t stream) {
  (void)in_sizes; (void)n_in; (void)out_size; (void)ws_size;

  const float* x   = (const float*)d_in[0];   // inputs
  const float* E   = (const float*)d_in[1];   // embeddings [D,K]
  const float* cse = (const float*)d_in[2];   // cluster_size_ema
  const float* ema = (const float*)d_in[3];   // embeddings_ema

  float* out      = (float*)d_out;
  float* q_out    = out;                                     // 16777216
  float* loss_out = out + (size_t)N_ROWS * EMB_D;            // 1
  float* nemb_out = loss_out + 1;                            // 524288
  float* ncse_out = nemb_out + (size_t)EMB_D * N_EMB;        // 1024
  float* nema_out = ncse_out + N_EMB;                        // 524288

  char* ws = (char*)d_ws;
  float*     enorm   = (float*)(ws + 0);            //   4 KB
  float*     counts  = (float*)(ws + 4096);         //   4 KB
  float*     cs      = (float*)(ws + 8192);         //   4 KB
  int*       idx     = (int*)  (ws + 16384);        // 128 KB
  float*     lossp   = (float*)(ws + 147456);       // 128 KB
  _Float16*  Bp      = (_Float16*)(ws + 294912);    //   1 MB packed codebook
  float*     emb_sum = (float*)(ws + 1572864);      //   2 MB

  hipMemsetAsync(counts,  0, N_EMB * sizeof(float), stream);
  hipMemsetAsync(emb_sum, 0, (size_t)EMB_D * N_EMB * sizeof(float), stream);

  vq_enorm_kernel <<<N_EMB / 256, 256, 0, stream>>>(E, enorm);
  vq_pack_b_kernel<<<32768 / 256,  256, 0, stream>>>(E, Bp);
  vq_argmin_gemm_kernel<<<N_ROWS / (16 * 8), 256, 0, stream>>>(x, Bp, enorm, idx);
  vq_quantize_scatter_kernel<<<N_ROWS, 512, 0, stream>>>(
      x, E, idx, q_out, emb_sum, counts, lossp);
  vq_finalize_kernel<<<1, 1024, 0, stream>>>(
      counts, cse, lossp, ncse_out, loss_out, cs);
  vq_emb_update_kernel<<<(EMB_D * N_EMB) / 256, 256, 0, stream>>>(
      ema, emb_sum, cs, nemb_out, nema_out);
}
